// SSM_Layer_60189671686224
// MI455X (gfx1250) — matrix operand order repaired and checked by reading, not compiled, not run on hardware
//
#include <hip/hip_runtime.h>
#include <hip/hip_bf16.h>

// ---------------------------------------------------------------------------
// Mamba SSM layer for MI455X (gfx1250), wave32 + WMMA bf16.
// Pipeline:
//   0) transpose_cast : x -> xT bf16 [M,K]; W_in/W_out -> [N,K] bf16
//   1) wmma_gemm      : xz  = xT @ W_in^T^T        (bf16 WMMA, f32 accum)
//   2) conv_silu      : xc  = silu(depthwise_conv(xi) + b)
//   3) proj_delta     : proj = xc@W_x ; delta = softplus(dt@W_dt + bias); B,C
//   4) scan           : selective scan, fused +xc*D_skip and *silu(z);
//                       emits bf16 activations for GEMM-2
//   5) wmma_gemm      : yW  = yact @ W_out
//   6) ln_transpose   : out = LN(yW + xt) transposed back to (B, DM, L)
// Workspace: ~341 MB scratch in d_ws.
// ---------------------------------------------------------------------------

#define B_   16
#define DM_  512
#define L_   1024
#define DI_  1024
#define DS_  16
#define DR_  32

typedef __attribute__((ext_vector_type(16))) __bf16 v16bf;
typedef __attribute__((ext_vector_type(8)))  __bf16 v8bf;
typedef __attribute__((ext_vector_type(8)))  float  v8f;

__device__ __forceinline__ __bf16 f2bf(float f) {
    unsigned int u = __builtin_bit_cast(unsigned int, f);
    unsigned int r = (u + 0x7FFFu + ((u >> 16) & 1u)) >> 16;   // RNE
    return __builtin_bit_cast(__bf16, (unsigned short)r);
}

// ---------------------------------------------------------------------------
// Tiled transpose + f32->bf16 cast: src (batch, R, C) f32 -> dst (batch, C, R)
// bf16. 32x32 tiles through LDS, coalesced on both sides.
// ---------------------------------------------------------------------------
__global__ __launch_bounds__(256)
void transpose_cast_kernel(const float* __restrict__ src, __bf16* __restrict__ dst,
                           int R, int C)
{
    __shared__ float t[32][33];
    const int tid = threadIdx.x;
    const int c0 = blockIdx.x * 32, r0 = blockIdx.y * 32;
    const size_t sb = (size_t)blockIdx.z * (size_t)R * (size_t)C;
#pragma unroll
    for (int i = 0; i < 4; ++i) {
        int idx = tid + i * 256;
        int rr = idx >> 5, cc = idx & 31;
        t[rr][cc] = src[sb + (size_t)(r0 + rr) * C + (c0 + cc)];
    }
    __syncthreads();
#pragma unroll
    for (int i = 0; i < 4; ++i) {
        int idx = tid + i * 256;
        int cc = idx >> 5, rr = idx & 31;
        dst[sb + (size_t)(c0 + cc) * R + (r0 + rr)] = f2bf(t[rr][cc]);
    }
}

// ---------------------------------------------------------------------------
// bf16 WMMA GEMM: C[M,N](f32) = A[M,K](bf16, row-major) * BT[N,K](bf16)^T.
// Block tile 128x128, K-chunk 32, 256 threads = 8 waves, one 16-row M-tile
// and 8 N-tiles per wave -> 8 v_wmma_f32_16x16x32_bf16 per K-step.
// LDS double-buffered; next K-tile prefetched into registers while the
// current tile's WMMAs run, then committed -> one barrier per K-step.
// ---------------------------------------------------------------------------
__global__ __launch_bounds__(256)
void wmma_gemm_bf16_kernel(const __bf16* __restrict__ A, const __bf16* __restrict__ BT,
                           float* __restrict__ C, int K, int N)
{
    __shared__ __align__(32) __bf16 sA[2][128 * 32];   // [m][k]  2 x 8 KB
    __shared__ __align__(32) __bf16 sB[2][128 * 32];   // [n][k]  2 x 8 KB

    const int tid  = threadIdx.x;
    const int lane = tid & 31;
    const int wv   = tid >> 5;          // wave id 0..7 -> M-tile row
    const int half = lane >> 4;
    const int mr   = lane & 15;
    const int m0 = blockIdx.y * 128;
    const int n0 = blockIdx.x * 128;

    // 16-byte copy chunks: chunk c -> row c>>2, k-offset (c&3)*8 (2 per thread)
    const int r0c = tid >> 2,        ko = (tid & 3) * 8;
    const int r1c = r0c + 64;
    const size_t aRow0 = (size_t)(m0 + r0c) * K + ko;
    const size_t aRow1 = (size_t)(m0 + r1c) * K + ko;
    const size_t bRow0 = (size_t)(n0 + r0c) * K + ko;
    const size_t bRow1 = (size_t)(n0 + r1c) * K + ko;
    const int lds0 = r0c * 32 + ko;
    const int lds1 = r1c * 32 + ko;

    v8f acc[8];
#pragma unroll
    for (int t = 0; t < 8; ++t)
        acc[t] = (v8f){0.f,0.f,0.f,0.f,0.f,0.f,0.f,0.f};

    // ---- preload K-tile 0 into buffer 0 ------------------------------------
    {
        uint4 a0 = *(const uint4*)&A[aRow0];
        uint4 a1 = *(const uint4*)&A[aRow1];
        uint4 b0 = *(const uint4*)&BT[bRow0];
        uint4 b1 = *(const uint4*)&BT[bRow1];
        *(uint4*)&sA[0][lds0] = a0;  *(uint4*)&sA[0][lds1] = a1;
        *(uint4*)&sB[0][lds0] = b0;  *(uint4*)&sB[0][lds1] = b1;
    }
    __syncthreads();

    int buf = 0;
    for (int k0 = 0; k0 < K; k0 += 32) {
        const int  nk  = k0 + 32;
        const bool has = nk < K;
        uint4 a0 = {0,0,0,0}, a1 = {0,0,0,0}, b0 = {0,0,0,0}, b1 = {0,0,0,0};
        if (has) {                       // prefetch next K-tile (global)
            a0 = *(const uint4*)&A[aRow0 + nk];
            a1 = *(const uint4*)&A[aRow1 + nk];
            b0 = *(const uint4*)&BT[bRow0 + nk];
            b1 = *(const uint4*)&BT[bRow1 + nk];
        }

        // A fragment: lane(m=mr, half) holds k = half*16 .. half*16+15
        v16bf af = *(const v16bf*)&sA[buf][(wv * 16 + mr) * 32 + half * 16];
#pragma unroll
        for (int t = 0; t < 8; ++t) {
            // B fragment: lane(n=mr, half) holds k = half*16 .. half*16+15
            v16bf bf_ = *(const v16bf*)&sB[buf][(t * 16 + mr) * 32 + half * 16];
            acc[t] = __builtin_amdgcn_wmma_f32_16x16x32_bf16(
                         false, af, false, bf_, (short)0, acc[t], false, false);
        }

        if (has) {                       // commit prefetch into other buffer
            *(uint4*)&sA[buf ^ 1][lds0] = a0;  *(uint4*)&sA[buf ^ 1][lds1] = a1;
            *(uint4*)&sB[buf ^ 1][lds0] = b0;  *(uint4*)&sB[buf ^ 1][lds1] = b1;
        }
        __syncthreads();
        buf ^= 1;
    }

    // ---- store: element r -> row m0 + wv*16 + half*8 + r, col n0 + t*16 + mr
#pragma unroll
    for (int t = 0; t < 8; ++t) {
        int n = n0 + t * 16 + mr;
#pragma unroll
        for (int r = 0; r < 8; ++r) {
            int m = m0 + wv * 16 + half * 8 + r;
            C[(size_t)m * N + n] = acc[t][r];
        }
    }
}

// ---------------------------------------------------------------------------
// Depthwise causal conv (D_CONV=4) + bias + SiLU.  xi = xz[:, 0:1024].
// ---------------------------------------------------------------------------
__global__ __launch_bounds__(256)
void conv_silu_kernel(const float* __restrict__ xz, const float* __restrict__ cw,
                      const float* __restrict__ cb, float* __restrict__ xc)
{
    const int row = blockIdx.x;          // b*L + l
    const int l   = row & (L_ - 1);
#pragma unroll
    for (int j = 0; j < 4; ++j) {
        int d = threadIdx.x + j * 256;
        float acc = cb[d];
#pragma unroll
        for (int k = 0; k < 4; ++k) {
            int lk = l + k - 3;
            if (lk >= 0)
                acc += xz[(size_t)(row + k - 3) * (2 * DI_) + d] * cw[d * 4 + k];
        }
        xc[(size_t)row * DI_ + d] = acc / (1.f + __expf(-acc));
    }
}

// ---------------------------------------------------------------------------
// proj = xc @ W_x (N=64); delta = softplus(proj[:,:32] @ W_dt + bias);
// B = proj[:,32:48]; C = proj[:,48:64].  One block per (b,l) row.
// ---------------------------------------------------------------------------
__global__ __launch_bounds__(256)
void proj_delta_kernel(const float* __restrict__ xc, const float* __restrict__ W_x,
                       const float* __restrict__ W_dt, const float* __restrict__ dt_bias,
                       float* __restrict__ delta, float* __restrict__ Bm,
                       float* __restrict__ Cm)
{
    __shared__ float sx[DI_];
    __shared__ float sp[256];
    __shared__ float sproj[64];
    const size_t row = blockIdx.x;
    const int tid = threadIdx.x;

#pragma unroll
    for (int j = 0; j < 4; ++j)
        sx[tid + j * 256] = xc[row * DI_ + tid + j * 256];
    __syncthreads();

    const int col = tid & 63, seg = tid >> 6;
    float a = 0.f;
    for (int k = seg * 256; k < seg * 256 + 256; ++k)
        a += sx[k] * W_x[(size_t)k * 64 + col];
    sp[tid] = a;
    __syncthreads();
    if (tid < 64)
        sproj[tid] = sp[tid] + sp[tid + 64] + sp[tid + 128] + sp[tid + 192];
    __syncthreads();

    if (tid < DS_)            Bm[row * DS_ + tid]         = sproj[DR_ + tid];
    else if (tid < 2 * DS_)   Cm[row * DS_ + (tid - DS_)] = sproj[DR_ + DS_ + (tid - DS_)];

#pragma unroll
    for (int j = 0; j < 4; ++j) {
        int d = tid + j * 256;
        float acc = dt_bias[d];
#pragma unroll
        for (int r = 0; r < DR_; ++r)
            acc += sproj[r] * W_dt[(size_t)r * DI_ + d];
        delta[row * DI_ + d] = (acc > 20.f) ? acc : log1pf(__expf(acc));
    }
}

// ---------------------------------------------------------------------------
// Selective scan, one thread per (b,d) channel, h[16] + A[16] in registers.
// B/C double-buffered in LDS; next-step delta/u/z register-prefetched so
// global latency overlaps the 16 v_exp_f32 chain (occupancy is ~1 wave/SIMD
// here, so software pipelining is the only latency hiding available).
// Fuses +xc*D_skip and *silu(z); emits bf16 for the GEMM-2 A operand.
// ---------------------------------------------------------------------------
__global__ __launch_bounds__(256)
void scan_kernel(const float* __restrict__ delta, const float* __restrict__ Bm,
                 const float* __restrict__ Cm, const float* __restrict__ xc,
                 const float* __restrict__ xz, const float* __restrict__ A_log,
                 const float* __restrict__ D_skip, __bf16* __restrict__ yact)
{
    const int b = blockIdx.x >> 2;                       // 64 blocks: 16 b * 4
    const int d = ((blockIdx.x & 3) << 8) + threadIdx.x;

    float Ad[DS_], h[DS_];
#pragma unroll
    for (int s = 0; s < DS_; ++s) {
        Ad[s] = -__expf(A_log[(size_t)d * DS_ + s]);
        h[s] = 0.f;
    }
    const float dskip = D_skip[d];

    __shared__ float sBC[2][2 * DS_];
    const size_t base = (size_t)b * L_;
    if (threadIdx.x < 2 * DS_)
        sBC[0][threadIdx.x] = (threadIdx.x < DS_)
            ? Bm[base * DS_ + threadIdx.x]
            : Cm[base * DS_ + (threadIdx.x - DS_)];
    __syncthreads();

    size_t row = base;
    float dl = delta[row * DI_ + d];
    float u  = xc[row * DI_ + d];
    float zv = xz[row * (2 * DI_) + DI_ + d];

    for (int l = 0; l < L_; ++l) {
        const float* bc = sBC[l & 1];
        if (threadIdx.x < 2 * DS_ && (l + 1) < L_) {
            size_t r1 = (base + l + 1) * DS_;
            sBC[(l + 1) & 1][threadIdx.x] = (threadIdx.x < DS_)
                ? Bm[r1 + threadIdx.x] : Cm[r1 + (threadIdx.x - DS_)];
        }
        float dl_n = 0.f, u_n = 0.f, zv_n = 0.f;
        if (l + 1 < L_) {
            size_t r1 = row + 1;
            dl_n = delta[r1 * DI_ + d];
            u_n  = xc[r1 * DI_ + d];
            zv_n = xz[r1 * (2 * DI_) + DI_ + d];
        }

        float y = 0.f;
        const float du = dl * u;
#pragma unroll
        for (int s = 0; s < DS_; ++s) {
            float dA = __expf(dl * Ad[s]);
            h[s] = dA * h[s] + du * bc[s];
            y += h[s] * bc[DS_ + s];
        }
        float g = zv / (1.f + __expf(-zv));              // silu(z)
        yact[row * DI_ + d] = f2bf((y + u * dskip) * g);

        dl = dl_n; u = u_n; zv = zv_n;
        row++;
        __syncthreads();
    }
}

// ---------------------------------------------------------------------------
// Residual + LayerNorm(DM=512) + transpose back to (B, DM, L).
// ---------------------------------------------------------------------------
__global__ __launch_bounds__(256)
void ln_transpose_kernel(const float* __restrict__ yW, const float* __restrict__ x,
                         const float* __restrict__ gamma, const float* __restrict__ beta,
                         float* __restrict__ out)
{
    __shared__ float st[8 * DM_];                        // 16 KB
    const int tid = threadIdx.x, wv = tid >> 5, lane = tid & 31;
    const int row0 = blockIdx.x * 8;                     // b*L + l0
    const int b = row0 >> 10, l0 = row0 & (L_ - 1);

    for (int d = tid; d < DM_; d += 256) {
        const float4* src = (const float4*)&x[((size_t)b * DM_ + d) * L_ + l0];
        float4 r0 = src[0], r1 = src[1];
        st[0 * DM_ + d] = r0.x; st[1 * DM_ + d] = r0.y;
        st[2 * DM_ + d] = r0.z; st[3 * DM_ + d] = r0.w;
        st[4 * DM_ + d] = r1.x; st[5 * DM_ + d] = r1.y;
        st[6 * DM_ + d] = r1.z; st[7 * DM_ + d] = r1.w;
    }
    __syncthreads();

    const size_t row = row0 + wv;
    float v[16], s = 0.f, s2 = 0.f;
#pragma unroll
    for (int i = 0; i < 16; ++i) {
        int d = lane + i * 32;
        float val = yW[row * DM_ + d] + st[wv * DM_ + d];
        v[i] = val; s += val; s2 += val * val;
    }
#pragma unroll
    for (int off = 16; off; off >>= 1) {
        s  += __shfl_xor(s,  off, 32);
        s2 += __shfl_xor(s2, off, 32);
    }
    const float mu   = s * (1.0f / DM_);
    const float var  = s2 * (1.0f / DM_) - mu * mu;
    const float rstd = rsqrtf(var + 1e-5f);
#pragma unroll
    for (int i = 0; i < 16; ++i) {
        int d = lane + i * 32;
        st[wv * DM_ + d] = (v[i] - mu) * rstd * gamma[d] + beta[d];
    }
    __syncthreads();

    for (int d = tid; d < DM_; d += 256) {
        float4 o0, o1;
        o0.x = st[0 * DM_ + d]; o0.y = st[1 * DM_ + d];
        o0.z = st[2 * DM_ + d]; o0.w = st[3 * DM_ + d];
        o1.x = st[4 * DM_ + d]; o1.y = st[5 * DM_ + d];
        o1.z = st[6 * DM_ + d]; o1.w = st[7 * DM_ + d];
        float4* dst = (float4*)&out[((size_t)b * DM_ + d) * L_ + l0];
        dst[0] = o0; dst[1] = o1;
    }
}

// ---------------------------------------------------------------------------
extern "C" void kernel_launch(void* const* d_in, const int* in_sizes, int n_in,
                              void* d_out, int out_size, void* d_ws, size_t ws_size,
                              hipStream_t stream) {
    const float* x       = (const float*)d_in[0];
    const float* W_in    = (const float*)d_in[1];
    const float* conv_w  = (const float*)d_in[2];
    const float* conv_b  = (const float*)d_in[3];
    const float* W_x     = (const float*)d_in[4];
    const float* W_dt    = (const float*)d_in[5];
    const float* dt_bias = (const float*)d_in[6];
    const float* A_log   = (const float*)d_in[7];
    const float* D_skip  = (const float*)d_in[8];
    const float* W_out   = (const float*)d_in[9];
    const float* gamma   = (const float*)d_in[10];
    const float* beta    = (const float*)d_in[11];
    float* out = (float*)d_out;

    const size_t M = (size_t)B_ * L_;                    // 16384
    char* p = (char*)d_ws;
    float*  xz      = (float*)p;   p += M * 2 * DI_ * sizeof(float);   // 128 MB
    float*  xc      = (float*)p;   p += M * DI_ * sizeof(float);       //  64 MB
    float*  delta   = (float*)p;   p += M * DI_ * sizeof(float);       //  64 MB
    float*  Bm      = (float*)p;   p += M * DS_ * sizeof(float);       //   1 MB
    float*  Cm      = (float*)p;   p += M * DS_ * sizeof(float);       //   1 MB
    float*  yW      = (float*)p;   p += M * DM_ * sizeof(float);       //  32 MB
    __bf16* xT      = (__bf16*)p;  p += M * DM_ * sizeof(__bf16);      //  16 MB
    __bf16* W_inT   = (__bf16*)p;  p += (size_t)2 * DI_ * DM_ * sizeof(__bf16); // 2 MB
    __bf16* W_outT  = (__bf16*)p;  p += (size_t)DM_ * DI_ * sizeof(__bf16);     // 1 MB
    __bf16* yact    = (__bf16*)p;  p += M * DI_ * sizeof(__bf16);      //  32 MB

    // 0) operand conversion / layout: all GEMM inputs become bf16, K-contig.
    //    x (16,512,1024) -> xT (16384,512)
    transpose_cast_kernel<<<dim3(L_ / 32, DM_ / 32, B_), 256, 0, stream>>>(
        x, xT, DM_, L_);
    //    W_in (512,2048) -> W_inT (2048,512)
    transpose_cast_kernel<<<dim3(2 * DI_ / 32, DM_ / 32, 1), 256, 0, stream>>>(
        W_in, W_inT, DM_, 2 * DI_);
    //    W_out (1024,512) -> W_outT (512,1024)
    transpose_cast_kernel<<<dim3(DM_ / 32, DI_ / 32, 1), 256, 0, stream>>>(
        W_out, W_outT, DI_, DM_);

    // 1) xz = xt @ W_in   (M=16384, K=512, N=2048)
    wmma_gemm_bf16_kernel<<<dim3(2 * DI_ / 128, M / 128), 256, 0, stream>>>(
        xT, W_inT, xz, DM_, 2 * DI_);
    // 2) conv + SiLU
    conv_silu_kernel<<<(int)M, 256, 0, stream>>>(xz, conv_w, conv_b, xc);
    // 3) proj / delta / B / C
    proj_delta_kernel<<<(int)M, 256, 0, stream>>>(xc, W_x, W_dt, dt_bias,
                                                  delta, Bm, Cm);
    // 4) selective scan (fused skip + gate, bf16 out)
    scan_kernel<<<B_ * DI_ / 256, 256, 0, stream>>>(delta, Bm, Cm, xc, xz,
                                                    A_log, D_skip, yact);
    // 5) yW = yact @ W_out   (M=16384, K=1024, N=512)
    wmma_gemm_bf16_kernel<<<dim3(DM_ / 128, M / 128), 256, 0, stream>>>(
        yact, W_outT, yW, DI_, DM_);
    // 6) residual + LayerNorm + transpose to (B, DM, L)
    ln_transpose_kernel<<<(int)(M / 8), 256, 0, stream>>>(yW, x, gamma, beta, out);
}
